// CliffordAttention_15685220565648
// MI455X (gfx1250) — compile-verified, matrix operand order checked by reading, not run on hardware
//
#include <hip/hip_runtime.h>
#include <hip/hip_bf16.h>
#include <stdint.h>

// ---------------------------------------------------------------------------
// Clifford attention for MI455X (gfx1250, wave32, WMMA bf16 16x16x32).
// ---------------------------------------------------------------------------

typedef __attribute__((ext_vector_type(16))) __bf16 v16bf;
typedef __attribute__((ext_vector_type(8)))  float  v8f;

#define DEV __device__ __forceinline__

#define BATCH 4
#define LSEQ  1024
#define DMOD  4096
#define NHD   16
#define HDIM  256   // CD * NB

DEV unsigned short f2bf(float f) {
  union { float f; unsigned u; } c; c.f = f;
  unsigned u = c.u;
  return (unsigned short)((u + 0x7FFFu + ((u >> 16) & 1u)) >> 16);
}
DEV float bf2f(unsigned short h) {
  union { unsigned u; float f; } c; c.u = ((unsigned)h) << 16; return c.f;
}

// Load a 16x32 bf16 A/B fragment chunk pair for this lane:
// 8 halves at p[0..7] (K+0..7 for half 0 / K+8..15 for half 1) and
// 8 halves at p[16..23] (K+16.. / K+24..).  p must be 16B aligned.
DEV v16bf ldfrag(const unsigned short* p) {
  v16bf v;
  ((uint4*)&v)[0] = ((const uint4*)p)[0];
  ((uint4*)&v)[1] = ((const uint4*)p)[2];
  return v;
}

DEV v8f wmma_bf16(v16bf a, v16bf b, v8f c) {
  return __builtin_amdgcn_wmma_f32_16x16x32_bf16(
      /*neg_a=*/false, a, /*neg_b=*/false, b,
      /*c_mod=*/(short)0, c, /*reuse_a=*/false, /*reuse_b=*/false);
}

// ---------------------------------------------------------------------------
// Kernel 0: fp32 -> bf16 cast
// ---------------------------------------------------------------------------
__global__ void __launch_bounds__(256)
cast_f32_bf16(const float* __restrict__ src, unsigned short* __restrict__ dst, int n) {
  int i = blockIdx.x * blockDim.x + threadIdx.x;
  if (i < n) dst[i] = f2bf(src[i]);
}

// ---------------------------------------------------------------------------
// Kernel 1: fused QKV projection GEMM.
//   C[m, n] = sum_k xb[m, k] * W[n, k] + bias[n],  m in [0, B*L), n in [0, DM)
// blockIdx.z selects Q/K/V.  Block: 256 threads = 8 waves, tile 64(M)x256(N),
// each wave computes a 32x64 subtile as 2x4 WMMA accumulators
// (8 WMMAs per 6 fragment loads per K-step).
// Q/K stored bf16 as [b, h, l, 256]; V stored transposed [b, h, c, m].
// ---------------------------------------------------------------------------
__global__ void __launch_bounds__(256)
qkv_gemm(const unsigned short* __restrict__ xb,
         const unsigned short* __restrict__ wqb,
         const unsigned short* __restrict__ wkb,
         const unsigned short* __restrict__ wvb,
         const float* __restrict__ bq, const float* __restrict__ bk,
         const float* __restrict__ bv,
         unsigned short* __restrict__ qo, unsigned short* __restrict__ ko,
         unsigned short* __restrict__ vo) {
  const int z = blockIdx.z;
  const unsigned short* wb   = (z == 0) ? wqb : (z == 1) ? wkb : wvb;
  const float*          bias = (z == 0) ? bq  : (z == 1) ? bk  : bv;
  unsigned short*       outp = (z == 0) ? qo  : (z == 1) ? ko  : vo;

  const int lane = threadIdx.x & 31, wid = threadIdx.x >> 5;
  const int hs = lane >> 4, ln = lane & 15;
  const int wm = wid & 1, wn = wid >> 1;
  const int mBase = blockIdx.y * 64 + wm * 32;
  const int nBase = blockIdx.x * 256 + wn * 64;

  // Per-lane row pointers (advance by K inside the loop).
  const unsigned short* pa[2];
  const unsigned short* pb[4];
#pragma unroll
  for (int i = 0; i < 2; i++)
    pa[i] = xb + (size_t)(mBase + i * 16 + ln) * DMOD + hs * 8;
#pragma unroll
  for (int j = 0; j < 4; j++)
    pb[j] = wb + (size_t)(nBase + j * 16 + ln) * DMOD + hs * 8;

  v8f acc[2][4];
#pragma unroll
  for (int i = 0; i < 2; i++)
#pragma unroll
    for (int j = 0; j < 4; j++)
#pragma unroll
      for (int e = 0; e < 8; e++) acc[i][j][e] = 0.f;

  for (int k0 = 0; k0 < DMOD; k0 += 32) {
    v16bf a[2], bfr[4];
#pragma unroll
    for (int i = 0; i < 2; i++) {
      a[i] = ldfrag(pa[i] + k0);
      __builtin_prefetch(pa[i] + k0 + 64, 0, 1);  // speculative, always safe
    }
#pragma unroll
    for (int j = 0; j < 4; j++) {
      bfr[j] = ldfrag(pb[j] + k0);
      __builtin_prefetch(pb[j] + k0 + 64, 0, 1);
    }
#pragma unroll
    for (int i = 0; i < 2; i++)
#pragma unroll
      for (int j = 0; j < 4; j++)
        acc[i][j] = wmma_bf16(a[i], bfr[j], acc[i][j]);
  }

  // C layout: VGPR r holds (row = base + r + hs*8, col = base + ln).
#pragma unroll
  for (int i = 0; i < 2; i++)
#pragma unroll
    for (int j = 0; j < 4; j++) {
#pragma unroll
      for (int r = 0; r < 8; r++) {
        int row = mBase + i * 16 + r + hs * 8;   // global token index
        int col = nBase + j * 16 + ln;           // output feature
        float v = acc[i][j][r] + bias[col];
        int b = row >> 10, l = row & 1023;
        int h = col >> 8,  c = col & 255;
        size_t idx = (z == 2)
            ? ((((size_t)(b * NHD + h)) * HDIM + c) * LSEQ + l)   // V^T
            : ((((size_t)(b * NHD + h)) * LSEQ + l) * HDIM + c);  // Q, K
        outp[idx] = f2bf(v);
      }
    }
}

// ---------------------------------------------------------------------------
// Kernel 2: fold reverse signs, logits scale and orientation-bias kernel into
// a per-head 8x8 blade transform of Q (in place):
//   qhat_j = sum_i q_i * rev_i * (scale*delta_ij + wk[h,i,j])
//   wk[h,i,j] = sum_c C[i,j,biv_c] * bias_weight[h,c]
// Structure constants hard-coded for g=(1,1,1), grade-sorted blade order.
// ---------------------------------------------------------------------------
__global__ void __launch_bounds__(256)
qhat_transform(unsigned short* __restrict__ q, const float* __restrict__ bw) {
  __shared__ float A[8][8];
  const int bh = blockIdx.x;        // b*16 + h
  const int h = bh & 15;
  const int t = threadIdx.x;
  if (t < 64) {
    const int i = t >> 3, j = t & 7;
    // 24 nonzero C[i,j,c] entries (c: 0=e12, 1=e13, 2=e23)
    const int   TI[24] = {0,4,1,2,3,7,5,6,  0,5,1,3,2,7,4,6,  0,6,2,3,1,7,4,5};
    const int   TJ[24] = {4,0,2,1,7,3,6,5,  5,0,3,1,7,2,6,4,  6,0,3,2,7,1,5,4};
    const float TS[24] = {1,1,1,-1,1,1,-1,1, 1,1,1,-1,-1,-1,1,-1, 1,1,1,-1,1,1,-1,1};
    float a = (i == j) ? 0.0625f : 0.0f;  // scale = 1/sqrt(256)
#pragma unroll
    for (int e = 0; e < 24; e++)
      if (TI[e] == i && TJ[e] == j) a += TS[e] * bw[h * 3 + e / 8];
    const float rev = (i >= 4) ? -1.f : 1.f;
    A[i][j] = rev * a;
  }
  __syncthreads();

  unsigned short* base = q + (size_t)bh * LSEQ * HDIM;
  for (int it = threadIdx.x; it < LSEQ * 32; it += blockDim.x) {
    unsigned short* p = base + (size_t)it * 8;
    float qi[8], qo[8];
#pragma unroll
    for (int i = 0; i < 8; i++) qi[i] = bf2f(p[i]);
#pragma unroll
    for (int j = 0; j < 8; j++) {
      float s = 0.f;
#pragma unroll
      for (int i = 0; i < 8; i++) s += qi[i] * A[i][j];
      qo[j] = s;
    }
#pragma unroll
    for (int j = 0; j < 8; j++) p[j] = f2bf(qo[j]);
  }
}

// ---------------------------------------------------------------------------
// Kernel 3: flash attention per (b, h).  Block = 128 threads = 4 waves; each
// wave owns 16 query rows.  Online softmax over m-chunks of 32.  P tiles are
// transposed C-layout -> A-layout through a per-wave LDS staging buffer
// (same-wave DS ordering; compiler inserts s_wait_dscnt, no barrier needed).
// ---------------------------------------------------------------------------
__global__ void __launch_bounds__(128)
attn_kernel(const unsigned short* __restrict__ qhat,
            const unsigned short* __restrict__ kb,
            const unsigned short* __restrict__ vt,
            float* __restrict__ out) {
  __shared__ __align__(16) unsigned short ps[4][16 * 32];

  const int lane = threadIdx.x & 31, wid = threadIdx.x >> 5;
  const int hs = lane >> 4, ln = lane & 15;
  const int bh = blockIdx.y;               // b*16 + h
  const int b = bh >> 4, h = bh & 15;
  const int l0 = blockIdx.x * 64 + wid * 16;

  const unsigned short* qbase = qhat + (size_t)bh * LSEQ * HDIM;
  const unsigned short* kbase = kb   + (size_t)bh * LSEQ * HDIM;
  const unsigned short* vbase = vt   + (size_t)bh * HDIM * LSEQ;

  // Preload all 8 A-fragments of qhat for this wave's 16 rows (K = 256).
  v16bf qh[8];
#pragma unroll
  for (int t = 0; t < 8; t++)
    qh[t] = ldfrag(qbase + (size_t)(l0 + ln) * HDIM + t * 32 + hs * 8);

  v8f o[16];
#pragma unroll
  for (int t = 0; t < 16; t++)
#pragma unroll
    for (int e = 0; e < 8; e++) o[t][e] = 0.f;

  float mrow[8], lrow[8];
#pragma unroll
  for (int r = 0; r < 8; r++) { mrow[r] = -1e30f; lrow[r] = 0.f; }

  unsigned short* pw = ps[wid];

  for (int mb = 0; mb < LSEQ; mb += 32) {
    // --- scores: two 16x16 tiles over K = 256 (8 WMMA steps each) ---
    v8f s0, s1;
#pragma unroll
    for (int e = 0; e < 8; e++) { s0[e] = 0.f; s1[e] = 0.f; }
#pragma unroll
    for (int t = 0; t < 8; t++) {
      v16bf b0 = ldfrag(kbase + (size_t)(mb +      ln) * HDIM + t * 32 + hs * 8);
      v16bf b1 = ldfrag(kbase + (size_t)(mb + 16 + ln) * HDIM + t * 32 + hs * 8);
      s0 = wmma_bf16(qh[t], b0, s0);
      s1 = wmma_bf16(qh[t], b1, s1);
    }

    // --- online softmax; P -> LDS as bf16 ---
#pragma unroll
    for (int r = 0; r < 8; r++) {
      float a0 = s0[r], a1 = s1[r];
      float mx = fmaxf(a0, a1);
#pragma unroll
      for (int d = 1; d < 16; d <<= 1) mx = fmaxf(mx, __shfl_xor(mx, d, 32));
      float mnew  = fmaxf(mrow[r], mx);
      float alpha = __expf(mrow[r] - mnew);
      float e0 = __expf(a0 - mnew), e1 = __expf(a1 - mnew);
      float rs = e0 + e1;
#pragma unroll
      for (int d = 1; d < 16; d <<= 1) rs += __shfl_xor(rs, d, 32);
      lrow[r] = lrow[r] * alpha + rs;
      mrow[r] = mnew;
#pragma unroll
      for (int t = 0; t < 16; t++) o[t][r] *= alpha;
      int row = r + hs * 8;
      pw[row * 32 + ln]      = f2bf(e0);
      pw[row * 32 + 16 + ln] = f2bf(e1);
    }

    // --- P (16x32) as A fragment from LDS; O += P @ V over 16 N-tiles ---
    v16bf pa = ldfrag(pw + ln * 32 + hs * 8);
#pragma unroll
    for (int t = 0; t < 16; t++) {
      v16bf bv =
          ldfrag(vbase + (size_t)(t * 16 + ln) * LSEQ + mb + hs * 8);
      o[t] = wmma_bf16(pa, bv, o[t]);
    }
  }

  // --- normalize (per-row reciprocal) and store fp32 output [B, L, DM] ---
  float rinv[8];
#pragma unroll
  for (int r = 0; r < 8; r++) rinv[r] = 1.0f / lrow[r];
#pragma unroll
  for (int t = 0; t < 16; t++)
#pragma unroll
    for (int r = 0; r < 8; r++) {
      int row = l0 + r + hs * 8;
      int c = t * 16 + ln;
      out[((size_t)b * LSEQ + row) * DMOD + h * HDIM + c] = o[t][r] * rinv[r];
    }
}

// ---------------------------------------------------------------------------
// Launch
// ---------------------------------------------------------------------------
extern "C" void kernel_launch(void* const* d_in, const int* in_sizes, int n_in,
                              void* d_out, int out_size, void* d_ws,
                              size_t ws_size, hipStream_t stream) {
  const float* x  = (const float*)d_in[0];
  const float* Wq = (const float*)d_in[1];
  const float* bq = (const float*)d_in[2];
  const float* Wk = (const float*)d_in[3];
  const float* bk = (const float*)d_in[4];
  const float* Wv = (const float*)d_in[5];
  const float* bv = (const float*)d_in[6];
  const float* bw = (const float*)d_in[7];
  float* out = (float*)d_out;

  char* ws = (char*)d_ws;
  const size_t SZ = (size_t)DMOD * DMOD * 2;  // 32 MB per bf16 buffer
  unsigned short* xb  = (unsigned short*)(ws);
  unsigned short* wqb = (unsigned short*)(ws + 1 * SZ);
  unsigned short* wkb = (unsigned short*)(ws + 2 * SZ);
  unsigned short* wvb = (unsigned short*)(ws + 3 * SZ);
  unsigned short* qb  = (unsigned short*)(ws + 4 * SZ);  // Q, then qhat in place
  unsigned short* kbf = (unsigned short*)(ws + 5 * SZ);
  unsigned short* vtb = (unsigned short*)(ws + 6 * SZ);

  const int NX = BATCH * LSEQ * DMOD;  // 16.7M
  const int NW = DMOD * DMOD;          // 16.7M
  cast_f32_bf16<<<(NX + 255) / 256, 256, 0, stream>>>(x, xb, NX);
  cast_f32_bf16<<<(NW + 255) / 256, 256, 0, stream>>>(Wq, wqb, NW);
  cast_f32_bf16<<<(NW + 255) / 256, 256, 0, stream>>>(Wk, wkb, NW);
  cast_f32_bf16<<<(NW + 255) / 256, 256, 0, stream>>>(Wv, wvb, NW);

  qkv_gemm<<<dim3(DMOD / 256, (BATCH * LSEQ) / 64, 3), 256, 0, stream>>>(
      xb, wqb, wkb, wvb, bq, bk, bv, qb, kbf, vtb);

  qhat_transform<<<dim3(BATCH * NHD), 256, 0, stream>>>(qb, bw);

  attn_kernel<<<dim3(LSEQ / 64, BATCH * NHD), 128, 0, stream>>>(
      qb, kbf, vtb, out);
}
